// EdgeUpdate_44590350467103
// MI455X (gfx1250) — compile-verified, matrix operand order checked by reading, not run on hardware
//
#include <hip/hip_runtime.h>
#include <hip/hip_bf16.h>

// CDNA5 / gfx1250 edge-MLP with bf16 WMMA (f32 accumulate), wave32.
// Each wave computes 32 edges (two 16-row M-tiles) so every B fragment
// (2x ds_load_b128) feeds two v_wmma_f32_16x16x32_bf16 instructions.

typedef __attribute__((ext_vector_type(16))) __bf16       v16bf;
typedef __attribute__((ext_vector_type(8)))  float        v8f;
typedef __attribute__((ext_vector_type(4)))  unsigned int v4u;

#if defined(__has_builtin)
#if __has_builtin(__builtin_amdgcn_cvt_pk_bf16_f32)
#define HAVE_CVT_PK_BF16 1
#endif
#endif

namespace {

constexpr int D_  = 64;
constexpr int R_  = 64;
constexpr int IN_ = 2 * D_ + R_;  // 192
constexpr int H_  = 128;

// LDS strides in bf16 elements (padded; rows stay 16B aligned)
constexpr int A_STRIDE  = IN_ + 8;  // 200 -> 400B rows
constexpr int H_STRIDE  = H_  + 8;  // 136 -> 272B rows
constexpr int W1_STRIDE = IN_ + 8;  // transposed W1 row = one output column
constexpr int W2_STRIDE = H_  + 8;  // transposed W2 row

constexpr int WAVES   = 4;
constexpr int THREADS = WAVES * 32;
constexpr int M_WAVE  = 32;               // edges per wave (2 M-tiles)
constexpr int TILE_E  = WAVES * M_WAVE;   // 128 edges / block

constexpr size_t OFF_W1T = 0;
constexpr size_t OFF_W2T = OFF_W1T + (size_t)H_ * W1_STRIDE * 2;           // 51200
constexpr size_t OFF_B1  = OFF_W2T + (size_t)R_ * W2_STRIDE * 2;           // +17408
constexpr size_t OFF_B2  = OFF_B1  + (size_t)H_ * 4;
constexpr size_t OFF_A   = OFF_B2  + (size_t)R_ * 4;
constexpr size_t OFF_HT  = OFF_A   + (size_t)WAVES * M_WAVE * A_STRIDE * 2;
constexpr size_t SMEM_BYTES = OFF_HT + (size_t)WAVES * M_WAVE * H_STRIDE * 2; // ~152 KB

union Frag {
  v4u   q[2];
  v16bf v;
};
static_assert(sizeof(Frag) == 32, "frag must be 8 VGPRs");

__device__ __forceinline__ unsigned short f2bf(float f) {
  union { float f; unsigned u; } x;
  x.f = f;
  unsigned r = x.u + 0x7FFFu + ((x.u >> 16) & 1u);  // round-to-nearest-even
  return (unsigned short)(r >> 16);
}

// Pack two f32 -> one dword of two bf16 (hardware v_cvt_pk_bf16_f32 if present)
__device__ __forceinline__ unsigned pkbf(float a, float b) {
#ifdef HAVE_CVT_PK_BF16
  typedef __attribute__((ext_vector_type(2))) __bf16 v2bf;
  union { v2bf v; unsigned u; } c;
  c.v = __builtin_amdgcn_cvt_pk_bf16_f32(a, b);
  return c.u;
#else
  return (unsigned)f2bf(a) | ((unsigned)f2bf(b) << 16);
#endif
}

}  // namespace

__global__ __launch_bounds__(THREADS)
void edge_mlp_wmma_bf16(const float* __restrict__ node,
                        const float* __restrict__ edge,
                        const int*   __restrict__ sidx,
                        const int*   __restrict__ didx,
                        const float* __restrict__ W1,
                        const float* __restrict__ b1,
                        const float* __restrict__ W2,
                        const float* __restrict__ b2,
                        float* __restrict__ out,
                        int E) {
  extern __shared__ char smem[];
  unsigned short* w1t = (unsigned short*)(smem + OFF_W1T);
  unsigned short* w2t = (unsigned short*)(smem + OFF_W2T);
  float*          b1s = (float*)(smem + OFF_B1);
  float*          b2s = (float*)(smem + OFF_B2);

  const int tid = threadIdx.x;

  // ---- Phase 0: stage weights fp32 -> bf16, transposed, into LDS ----------
  for (int i = tid; i < IN_ * H_; i += THREADS) {
    int k = i >> 7;          // W1 row (K)
    int n = i & (H_ - 1);    // W1 col (N) -> consecutive lanes read consecutive n
    w1t[n * W1_STRIDE + k] = f2bf(W1[i]);
  }
  for (int i = tid; i < H_ * R_; i += THREADS) {
    int k = i >> 6;
    int n = i & (R_ - 1);
    w2t[n * W2_STRIDE + k] = f2bf(W2[i]);
  }
  if (tid < H_) b1s[tid] = b1[tid];
  if (tid < R_) b2s[tid] = b2[tid];
  __syncthreads();

  const int wave = tid >> 5;
  const int lane = tid & 31;
  const int tileBase = blockIdx.x * TILE_E + wave * M_WAVE;
  if (tileBase >= E) return;

  unsigned short* aT = (unsigned short*)(smem + OFF_A)  + (size_t)wave * M_WAVE * A_STRIDE;
  unsigned short* hT = (unsigned short*)(smem + OFF_HT) + (size_t)wave * M_WAVE * H_STRIDE;

  const int m    = lane & 15;        // row/col within a 16x16 tile
  const int koff = (lane >> 4) * 8;  // K sub-offset per ISA A/B lane layout
  const int mB   = (lane >> 4) * 8;  // C layout: VGPR v -> M = v + mB

  // ---- Phase 1: gather [src | dst | rbf] -> bf16 A tile [32 x 192] --------
  // Perfect lane<->row mapping: lane l owns edge row l. Fully unrolled 48-chunk
  // loop: segment select is compile-time, 2 float4 loads -> 1 ds_store_b128.
  {
    int  er    = tileBase + lane;
    bool valid = er < E;
    int  erc   = valid ? er : (E - 1);
    const float4* p0 = (const float4*)node + (size_t)sidx[erc] * 16;  // src row
    const float4* p1 = (const float4*)node + (size_t)didx[erc] * 16;  // dst row
    const float4* p2 = (const float4*)edge + (size_t)erc * 16;        // rbf row
    unsigned short* aRow = aT + lane * A_STRIDE;

#pragma unroll
    for (int c = 0; c < 48; c += 2) {
      const float4* p = (c < 16) ? (p0 + c)
                      : (c < 32) ? (p1 + (c - 16))
                                 : (p2 + (c - 32));
      float4 x = valid ? p[0] : make_float4(0.f, 0.f, 0.f, 0.f);
      float4 y = valid ? p[1] : make_float4(0.f, 0.f, 0.f, 0.f);
      v4u pk = {pkbf(x.x, x.y), pkbf(x.z, x.w), pkbf(y.x, y.y), pkbf(y.z, y.w)};
      *(v4u*)(aRow + c * 4) = pk;  // ds_store_b128 (16B aligned: c even)
    }
  }
  asm volatile("" ::: "memory");  // keep DS stores ordered before frag loads

  // ---- Layer 1: [32x192] @ [192x128]: 6 K-steps x 8 N x 2 M = 96 WMMA -----
  v8f acc[2][8];
#pragma unroll
  for (int t = 0; t < 2; ++t)
#pragma unroll
    for (int n = 0; n < 8; ++n)
      acc[t][n] = v8f{0.f, 0.f, 0.f, 0.f, 0.f, 0.f, 0.f, 0.f};

  const unsigned short* aRow0 = aT + m * A_STRIDE;
  const unsigned short* aRow1 = aT + (m + 16) * A_STRIDE;
#pragma unroll
  for (int kb = 0; kb < IN_ / 32; ++kb) {
    Frag a0, a1;
    const unsigned short* ap0 = aRow0 + kb * 32 + koff;
    const unsigned short* ap1 = aRow1 + kb * 32 + koff;
    a0.q[0] = *(const v4u*)(ap0);
    a0.q[1] = *(const v4u*)(ap0 + 16);
    a1.q[0] = *(const v4u*)(ap1);
    a1.q[1] = *(const v4u*)(ap1 + 16);
#pragma unroll
    for (int n = 0; n < 8; ++n) {
      Frag b;
      const unsigned short* bp = w1t + (n * 16 + m) * W1_STRIDE + kb * 32 + koff;
      b.q[0] = *(const v4u*)(bp);
      b.q[1] = *(const v4u*)(bp + 16);
      acc[0][n] = __builtin_amdgcn_wmma_f32_16x16x32_bf16(
          false, a0.v, false, b.v, (short)0, acc[0][n], false, false);
      acc[1][n] = __builtin_amdgcn_wmma_f32_16x16x32_bf16(
          false, a1.v, false, b.v, (short)0, acc[1][n], false, false);
    }
  }

  // ---- Bias + softplus, write bf16 hidden tile in A-operand layout --------
#pragma unroll
  for (int t = 0; t < 2; ++t) {
#pragma unroll
    for (int n = 0; n < 8; ++n) {
      int   col  = n * 16 + m;
      float bias = b1s[col];
#pragma unroll
      for (int v = 0; v < 8; ++v) {
        float xh = acc[t][n][v] + bias;
        float sp = fmaxf(xh, 0.f) + __logf(1.f + __expf(-fabsf(xh)));
        hT[(t * 16 + mB + v) * H_STRIDE + col] = f2bf(sp);
      }
    }
  }
  asm volatile("" ::: "memory");

  // ---- Layer 2: [32x128] @ [128x64]: 4 K x 4 N x 2 M = 32 WMMA ------------
  v8f acc2[2][4];
#pragma unroll
  for (int t = 0; t < 2; ++t)
#pragma unroll
    for (int n = 0; n < 4; ++n)
      acc2[t][n] = v8f{0.f, 0.f, 0.f, 0.f, 0.f, 0.f, 0.f, 0.f};

  const unsigned short* hRow0 = hT + m * H_STRIDE;
  const unsigned short* hRow1 = hT + (m + 16) * H_STRIDE;
#pragma unroll
  for (int kb = 0; kb < H_ / 32; ++kb) {
    Frag a0, a1;
    const unsigned short* ap0 = hRow0 + kb * 32 + koff;
    const unsigned short* ap1 = hRow1 + kb * 32 + koff;
    a0.q[0] = *(const v4u*)(ap0);
    a0.q[1] = *(const v4u*)(ap0 + 16);
    a1.q[0] = *(const v4u*)(ap1);
    a1.q[1] = *(const v4u*)(ap1 + 16);
#pragma unroll
    for (int n = 0; n < 4; ++n) {
      Frag b;
      const unsigned short* bp = w2t + (n * 16 + m) * W2_STRIDE + kb * 32 + koff;
      b.q[0] = *(const v4u*)(bp);
      b.q[1] = *(const v4u*)(bp + 16);
      acc2[0][n] = __builtin_amdgcn_wmma_f32_16x16x32_bf16(
          false, a0.v, false, b.v, (short)0, acc2[0][n], false, false);
      acc2[1][n] = __builtin_amdgcn_wmma_f32_16x16x32_bf16(
          false, a1.v, false, b.v, (short)0, acc2[1][n], false, false);
    }
  }

  // ---- Bias + store output (f32); lanes 0-15 cover contiguous 64B runs ----
#pragma unroll
  for (int t = 0; t < 2; ++t) {
#pragma unroll
    for (int n = 0; n < 4; ++n) {
      int   col  = n * 16 + m;
      float bias = b2s[col];
#pragma unroll
      for (int v = 0; v < 8; ++v) {
        int row = tileBase + t * 16 + mB + v;
        if (row < E) out[(size_t)row * R_ + col] = acc2[t][n][v] + bias;
      }
    }
  }
}

extern "C" void kernel_launch(void* const* d_in, const int* in_sizes, int n_in,
                              void* d_out, int out_size, void* d_ws, size_t ws_size,
                              hipStream_t stream) {
  const float* node = (const float*)d_in[0];
  const float* edge = (const float*)d_in[1];
  const int*   sidx = (const int*)d_in[2];
  const int*   didx = (const int*)d_in[3];
  const float* W1   = (const float*)d_in[4];
  const float* b1   = (const float*)d_in[5];
  const float* W2   = (const float*)d_in[6];
  const float* b2   = (const float*)d_in[7];
  float*       out  = (float*)d_out;

  const int E = in_sizes[2];  // src_idx count
  const int blocks = (E + TILE_E - 1) / TILE_E;

  // ~152 KB dynamic LDS (gfx1250 WGP has 320 KB); opt in above default cap.
  (void)hipFuncSetAttribute((const void*)edge_mlp_wmma_bf16,
                            hipFuncAttributeMaxDynamicSharedMemorySize,
                            (int)SMEM_BYTES);

  edge_mlp_wmma_bf16<<<blocks, THREADS, SMEM_BYTES, stream>>>(
      node, edge, sidx, didx, W1, b1, W2, b2, out, E);
}
